// DeepSetAttentionModel_91130616087191
// MI455X (gfx1250) — matrix operand order repaired and checked by reading, not compiled, not run on hardware
//
#include <hip/hip_runtime.h>

typedef _Float16 v16h __attribute__((ext_vector_type(16)));
typedef _Float16 v8h  __attribute__((ext_vector_type(8)));
typedef float    v8f  __attribute__((ext_vector_type(8)));

#define SEG_LEN 4097
#define NBATCH  64
#define TSEQ    4096

__device__ __forceinline__ v16h cat16(v8h lo, v8h hi) {
  return __builtin_shufflevector(lo, hi, 0,1,2,3,4,5,6,7,8,9,10,11,12,13,14,15);
}
__device__ __forceinline__ v8f wmma_f16(v16h a, v16h b, v8f c) {
  return __builtin_amdgcn_wmma_f32_16x16x32_f16(false, a, false, b, (short)0, c, false, false);
}

// ---------------------------------------------------------------------------
// K0: convert phi/psi weights f32 -> f16, pre-swizzled into WMMA B-tile order,
//     and fold U[h] = W_k[:, h*64:(h+1)*64] @ W_q[h]  (176 -> u48pad[4][64], u128[4][128])
// B tile layout (per 16x16xK=32 k-tile, col-tile): 512 halves; lane L holds
// col = L&15, halves i=0..15 -> K = kk*32 + (L>>4)*8 + (i&7) + (i>>3)*16, stored at L*16+i.
// ---------------------------------------------------------------------------
__global__ void prep_kernel(
    const float* pW0, const float* pW1, const float* pW2, const float* pW3,
    const float* qW0, const float* qW1, const float* qW2,
    const float* Wk, const float* Wq,
    _Float16* __restrict__ wAll, float* __restrict__ u48p, float* __restrict__ u128)
{
  const int offH[7] = {0, 8192, 24576, 40960, 57344, 65536, 81920};
  const int kdim[7] = {48, 128, 128, 128, 48, 128, 128};
  const float* srcs[7] = {pW0, pW1, pW2, pW3, qW0, qW1, qW2};
  for (int idx = blockIdx.x * blockDim.x + threadIdx.x; idx < 98304 + 768;
       idx += gridDim.x * blockDim.x) {
    if (idx < 98304) {
      int layer = 0;
      #pragma unroll
      for (int l = 1; l < 7; ++l) layer += (idx >= offH[l]) ? 1 : 0;
      int rel = idx - offH[layer];
      int tileI = rel >> 9, pos = rel & 511;
      int L = pos >> 4, i = pos & 15;
      int kk = tileI >> 3, j = tileI & 7;
      int K = kk * 32 + (L >> 4) * 8 + (i & 7) + (i >> 3) * 16;
      int col = j * 16 + (L & 15);
      float v = (K < kdim[layer]) ? srcs[layer][K * 128 + col] : 0.f;
      wAll[idx] = (_Float16)v;
    } else {
      int r = idx - 98304;
      if (r < 704) {
        int h = r / 176, k = r % 176;
        float s = 0.f;
        for (int d = 0; d < 64; ++d) s += Wk[k * 256 + h * 64 + d] * Wq[h * 64 + d];
        if (k < 48) u48p[h * 64 + k] = s;
        else        u128[h * 128 + (k - 48)] = s;
      } else {
        int rr = r - 704;
        u48p[(rr >> 4) * 64 + 48 + (rr & 15)] = 0.f;  // zero pad 48..63
      }
    }
  }
}

// ---------------------------------------------------------------------------
// K2: build the 48 (padded to 64) input features per time step (f16) and the
//     folded attention dot cdot[n][h] = collected[n] . u48[h].
// ---------------------------------------------------------------------------
__global__ void feat_kernel(
    const float* __restrict__ times, const float* __restrict__ values,
    const int* __restrict__ meas, const float* __restrict__ ts,
    const float* __restrict__ u48p,
    _Float16* __restrict__ feat, float* __restrict__ cdot)
{
  long i = (long)blockIdx.x * blockDim.x + threadIdx.x;
  if (i >= (long)NBATCH * TSEQ) return;
  int b = (int)(i >> 12), t = (int)(i & 4095);
  long n = (long)b * SEG_LEN + 1 + t;
  float tm = times[i], val = values[i];
  int m = meas[i];
  float f[11];
  #pragma unroll
  for (int k = 0; k < 5; ++k) {
    float x = tm / ts[k];
    f[k]     = sinf(x);
    f[5 + k] = cosf(x);
  }
  f[10] = val;
  v8h chunks[8];
  #pragma unroll
  for (int k = 0; k < 64; ++k) {
    float v;
    if (k < 11)       v = f[k];
    else if (k < 48)  v = ((k - 11) == m) ? 1.f : 0.f;
    else              v = 0.f;
    chunks[k >> 3][k & 7] = (_Float16)v;
  }
  v8h* fr = (v8h*)(feat + n * 64);
  #pragma unroll
  for (int q = 0; q < 8; ++q) fr[q] = chunks[q];
  #pragma unroll
  for (int h = 0; h < 4; ++h) {
    float c = u48p[h * 64 + 11 + m];
    #pragma unroll
    for (int k = 0; k < 11; ++k) c += f[k] * u48p[h * 64 + k];
    cdot[n * 4 + h] = c;
  }
}

// ---------------------------------------------------------------------------
// K1: demo rows (row 0 of each segment): demo MLP -> feat row, plus f32 phi /
//     psi / cdot for those 64 rows (negligible work; keeps main kernel uniform)
// ---------------------------------------------------------------------------
__global__ __launch_bounds__(128) void demo_kernel(
    const float* demo, const float* dW1, const float* db1,
    const float* dW2, const float* db2,
    const float* pW0, const float* pb0, const float* pW1, const float* pb1,
    const float* pW2, const float* pb2, const float* pW3, const float* pb3,
    const float* qW0, const float* qb0, const float* qW1, const float* qb1,
    const float* qW2, const float* qb2,
    const float* u48p,
    _Float16* feat, _Float16* encoded, float* cdot, float* segsum)
{
  int b = blockIdx.x, j = threadIdx.x;
  __shared__ float sd[8], sh[128], sde[64], sA[128], sB[128];
  if (j < 8) sd[j] = demo[b * 8 + j];
  __syncthreads();
  float acc = db1[j];
  for (int k = 0; k < 8; ++k) acc += sd[k] * dW1[k * 128 + j];
  sh[j] = fmaxf(acc, 0.f);
  __syncthreads();
  if (j < 64) {
    float v = 0.f;
    if (j < 48) {
      v = db2[j];
      for (int k = 0; k < 128; ++k) v += sh[k] * dW2[k * 48 + j];
    }
    sde[j] = v;
  }
  __syncthreads();
  long n0 = (long)b * SEG_LEN;
  if (j < 64) feat[n0 * 64 + j] = (_Float16)sde[j];
  if (j < 4) {
    float c = 0.f;
    for (int k = 0; k < 48; ++k) c += sde[k] * u48p[j * 64 + k];
    cdot[n0 * 4 + j] = c;
  }
  // phi (f32)
  acc = pb0[j]; for (int k = 0; k < 48;  ++k) acc += sde[k] * pW0[k * 128 + j];
  sA[j] = fmaxf(acc, 0.f); __syncthreads();
  acc = pb1[j]; for (int k = 0; k < 128; ++k) acc += sA[k] * pW1[k * 128 + j];
  sB[j] = fmaxf(acc, 0.f); __syncthreads();
  acc = pb2[j]; for (int k = 0; k < 128; ++k) acc += sB[k] * pW2[k * 128 + j];
  sA[j] = fmaxf(acc, 0.f); __syncthreads();
  acc = pb3[j]; for (int k = 0; k < 128; ++k) acc += sA[k] * pW3[k * 128 + j];
  encoded[n0 * 128 + j] = (_Float16)fmaxf(acc, 0.f);
  // psi (f32)
  acc = qb0[j]; for (int k = 0; k < 48;  ++k) acc += sde[k] * qW0[k * 128 + j];
  sB[j] = fmaxf(acc, 0.f); __syncthreads();
  acc = qb1[j]; for (int k = 0; k < 128; ++k) acc += sB[k] * qW1[k * 128 + j];
  sA[j] = fmaxf(acc, 0.f); __syncthreads();
  acc = qb2[j]; for (int k = 0; k < 128; ++k) acc += sA[k] * qW2[k * 128 + j];
  atomicAdd(segsum + b * 128 + j, fmaxf(acc, 0.f));
}

// ---------------------------------------------------------------------------
// Main WMMA kernel: per block = 128 rows of one segment (8 waves x 16 rows).
// All phi+psi weights staged in LDS (196608 B) + 4KB/wave activation scratch
// for the D-layout -> A-layout transpose between layers. 192 WMMAs per wave.
// ---------------------------------------------------------------------------
__global__ __launch_bounds__(256) void mlp_kernel(
    const _Float16* __restrict__ feat, const _Float16* __restrict__ wAllG,
    const float* __restrict__ phi_b0, const float* __restrict__ phi_b1,
    const float* __restrict__ phi_b2, const float* __restrict__ phi_b3,
    const float* __restrict__ psi_b0, const float* __restrict__ psi_b1,
    const float* __restrict__ psi_b2,
    _Float16* __restrict__ encoded, float* __restrict__ segsum)
{
  extern __shared__ char lds[];
  const int tid = threadIdx.x;
  { // stage weights (196608 B) into LDS
    uint4* d = (uint4*)lds;
    const uint4* s = (const uint4*)wAllG;
    for (int i = tid; i < 12288; i += 256) d[i] = s[i];
  }
  __syncthreads();

  const int wave = tid >> 5, lane = tid & 31;
  const int lrow = lane & 15, lgrp = lane >> 4;
  const int b = blockIdx.x >> 5;
  const int tile = blockIdx.x & 31;
  const long n0 = (long)b * SEG_LEN + 1 + tile * 128 + wave * 16;

  const _Float16* ldsW = (const _Float16*)lds;
  _Float16* act = (_Float16*)(lds + 196608 + wave * 4096);

  // A tiles of the input features (K=64), kept live for both phi and psi L0
  v16h fA0, fA1;
  {
    const _Float16* rp = feat + (n0 + lrow) * 64;
    fA0 = cat16(*(const v8h*)(rp + lgrp * 8),      *(const v8h*)(rp + 16 + lgrp * 8));
    fA1 = cat16(*(const v8h*)(rp + 32 + lgrp * 8), *(const v8h*)(rp + 48 + lgrp * 8));
  }

  v8f  accs[8];
  v16h a[4];

  auto loadB = [&](int wOffH, int kk, int j) -> v16h {
    const _Float16* p = ldsW + wOffH + ((kk * 8 + j) << 9) + lane * 16;
    return cat16(*(const v8h*)p, *(const v8h*)(p + 8));
  };
  auto layerK64 = [&](int wOffH) {
    #pragma unroll
    for (int j = 0; j < 8; ++j) {
      v8f c = {0.f, 0.f, 0.f, 0.f, 0.f, 0.f, 0.f, 0.f};
      c = wmma_f16(fA0, loadB(wOffH, 0, j), c);
      c = wmma_f16(fA1, loadB(wOffH, 1, j), c);
      accs[j] = c;
    }
  };
  auto layerK128 = [&](int wOffH) {
    #pragma unroll
    for (int j = 0; j < 8; ++j) {
      v8f c = {0.f, 0.f, 0.f, 0.f, 0.f, 0.f, 0.f, 0.f};
      #pragma unroll
      for (int kk = 0; kk < 4; ++kk) c = wmma_f16(a[kk], loadB(wOffH, kk, j), c);
      accs[j] = c;
    }
  };
  auto storeAct = [&](const float* bias) {  // bias + ReLU + D-layout -> [row][col] LDS
    #pragma unroll
    for (int j = 0; j < 8; ++j) {
      float bv = bias[j * 16 + lrow];
      _Float16* cp = act + lgrp * 8 * 128 + j * 16 + lrow;
      #pragma unroll
      for (int i = 0; i < 8; ++i) {
        float v = accs[j][i] + bv;
        cp[i * 128] = (_Float16)(v > 0.f ? v : 0.f);
      }
    }
  };
  auto loadActA = [&]() {  // [row][col] LDS -> A layout regs (K=128)
    #pragma unroll
    for (int kk = 0; kk < 4; ++kk) {
      const _Float16* p = act + lrow * 128 + kk * 32 + lgrp * 8;
      a[kk] = cat16(*(const v8h*)p, *(const v8h*)(p + 16));
    }
  };

  // phi: 48(pad64) -> 128 -> 128 -> 128 -> 128
  layerK64(0);       storeAct(phi_b0); loadActA();
  layerK128(8192);   storeAct(phi_b1); loadActA();
  layerK128(24576);  storeAct(phi_b2); loadActA();
  layerK128(40960);  storeAct(phi_b3);
  { // act (16 rows x 128 f16, already ReLU'd) -> encoded, coalesced
    uint4* gd = (uint4*)(encoded + n0 * 128);
    const uint4* ls = (const uint4*)act;
    #pragma unroll
    for (int i = 0; i < 8; ++i) gd[i * 32 + lane] = ls[i * 32 + lane];
  }
  // psi: 48(pad64) -> 128 -> 128 -> 128, then column-sum into segment sums
  layerK64(57344);   storeAct(psi_b0); loadActA();
  layerK128(65536);  storeAct(psi_b1); loadActA();
  layerK128(81920);
  #pragma unroll
  for (int j = 0; j < 8; ++j) {
    float bv = psi_b2[j * 16 + lrow];
    float s = 0.f;
    #pragma unroll
    for (int i = 0; i < 8; ++i) {
      float v = accs[j][i] + bv;
      s += (v > 0.f) ? v : 0.f;
    }
    s += __shfl_xor(s, 16, 32);
    if (lane < 16) atomicAdd(segsum + b * 128 + j * 16 + lane, s);
  }
}

// ---------------------------------------------------------------------------
// K3: aggdot[b][h] = (segsum[b]/4097) . u128[h]
// ---------------------------------------------------------------------------
__global__ void aggdot_kernel(const float* __restrict__ segsum,
                              const float* __restrict__ u128,
                              float* __restrict__ aggdot)
{
  int t = threadIdx.x;             // 256 = 64 segments x 4 heads
  int b = t >> 2, h = t & 3;
  float s = 0.f;
  for (int k = 0; k < 128; ++k) s += segsum[b * 128 + k] * u128[h * 128 + k];
  aggdot[b * 4 + h] = s * (1.0f / 4097.0f);
}

// ---------------------------------------------------------------------------
// K4: per-(segment,head) softmax stats: off = aggdot/8 - m, invs = 1/sum(exp)
// ---------------------------------------------------------------------------
__global__ __launch_bounds__(256) void softstat_kernel(
    const float* __restrict__ cdot, const float* __restrict__ aggdot,
    float* __restrict__ offv, float* __restrict__ invs)
{
  int b = blockIdx.x, tid = threadIdx.x;
  __shared__ float4 red[256];
  const float4* c4 = (const float4*)cdot + (long)b * SEG_LEN;
  float4 mx = make_float4(-1e30f, -1e30f, -1e30f, -1e30f);
  for (int i = tid; i < SEG_LEN; i += 256) {
    float4 v = c4[i];
    mx.x = fmaxf(mx.x, v.x); mx.y = fmaxf(mx.y, v.y);
    mx.z = fmaxf(mx.z, v.z); mx.w = fmaxf(mx.w, v.w);
  }
  red[tid] = mx; __syncthreads();
  for (int s = 128; s > 0; s >>= 1) {
    if (tid < s) {
      float4 o = red[tid + s];
      red[tid].x = fmaxf(red[tid].x, o.x); red[tid].y = fmaxf(red[tid].y, o.y);
      red[tid].z = fmaxf(red[tid].z, o.z); red[tid].w = fmaxf(red[tid].w, o.w);
    }
    __syncthreads();
  }
  float4 cm = red[0];
  float4 ad = ((const float4*)aggdot)[b];
  float m0 = (cm.x + ad.x) * 0.125f, m1 = (cm.y + ad.y) * 0.125f;
  float m2 = (cm.z + ad.z) * 0.125f, m3 = (cm.w + ad.w) * 0.125f;
  __syncthreads();
  float4 sm = make_float4(0.f, 0.f, 0.f, 0.f);
  for (int i = tid; i < SEG_LEN; i += 256) {
    float4 v = c4[i];
    sm.x += __expf((v.x + ad.x) * 0.125f - m0);
    sm.y += __expf((v.y + ad.y) * 0.125f - m1);
    sm.z += __expf((v.z + ad.z) * 0.125f - m2);
    sm.w += __expf((v.w + ad.w) * 0.125f - m3);
  }
  red[tid] = sm; __syncthreads();
  for (int s = 128; s > 0; s >>= 1) {
    if (tid < s) {
      float4 o = red[tid + s];
      red[tid].x += o.x; red[tid].y += o.y; red[tid].z += o.z; red[tid].w += o.w;
    }
    __syncthreads();
  }
  if (tid == 0) {
    float4 tot = red[0];
    offv[b * 4 + 0] = ad.x * 0.125f - m0; invs[b * 4 + 0] = 1.f / tot.x;
    offv[b * 4 + 1] = ad.y * 0.125f - m1; invs[b * 4 + 1] = 1.f / tot.y;
    offv[b * 4 + 2] = ad.z * 0.125f - m2; invs[b * 4 + 2] = 1.f / tot.z;
    offv[b * 4 + 3] = ad.w * 0.125f - m3; invs[b * 4 + 3] = 1.f / tot.w;
  }
}

// ---------------------------------------------------------------------------
// K5: agg_out[b][h*128+j] = sum_n attn[n,h] * encoded[n,j]
// ---------------------------------------------------------------------------
__global__ __launch_bounds__(256) void aggregate_kernel(
    const _Float16* __restrict__ encoded, const float* __restrict__ cdot,
    const float* __restrict__ offv, const float* __restrict__ invs,
    float* __restrict__ aggout)
{
  int b = blockIdx.x >> 3, chunk = blockIdx.x & 7;
  int t = threadIdx.x;
  int j = t & 127, pair = t >> 7;
  long nbeg = (long)b * SEG_LEN + chunk * 512;
  long nend = (chunk == 7) ? (long)(b + 1) * SEG_LEN : nbeg + 512;
  float o0 = offv[b * 4 + 2 * pair],     o1 = offv[b * 4 + 2 * pair + 1];
  float s0 = invs[b * 4 + 2 * pair],     s1 = invs[b * 4 + 2 * pair + 1];
  float acc0 = 0.f, acc1 = 0.f;
  for (long n = nbeg; n < nend; ++n) {
    float e = (float)encoded[n * 128 + j];
    float2 c = *(const float2*)(cdot + n * 4 + 2 * pair);
    acc0 += __expf(c.x * 0.125f + o0) * e;
    acc1 += __expf(c.y * 0.125f + o1) * e;
  }
  atomicAdd(aggout + b * 512 + (2 * pair) * 128 + j,     acc0 * s0);
  atomicAdd(aggout + b * 512 + (2 * pair + 1) * 128 + j, acc1 * s1);
}

// ---------------------------------------------------------------------------
// K6: rho MLP 512 -> 128 -> 128 -> 128 -> 1 + sigmoid
// ---------------------------------------------------------------------------
__global__ __launch_bounds__(128) void rho_kernel(
    const float* __restrict__ aggout,
    const float* rW0, const float* rb0, const float* rW1, const float* rb1,
    const float* rW2, const float* rb2, const float* rW3, const float* rb3,
    float* __restrict__ out)
{
  int b = blockIdx.x, j = threadIdx.x;
  __shared__ float sin_[512], sA[128], sB[128], red[128];
  for (int i = j; i < 512; i += 128) sin_[i] = aggout[b * 512 + i];
  __syncthreads();
  float acc = rb0[j];
  for (int k = 0; k < 512; ++k) acc += sin_[k] * rW0[k * 128 + j];
  sA[j] = fmaxf(acc, 0.f); __syncthreads();
  acc = rb1[j];
  for (int k = 0; k < 128; ++k) acc += sA[k] * rW1[k * 128 + j];
  sB[j] = fmaxf(acc, 0.f); __syncthreads();
  acc = rb2[j];
  for (int k = 0; k < 128; ++k) acc += sB[k] * rW2[k * 128 + j];
  float h3 = fmaxf(acc, 0.f);
  red[j] = h3 * rW3[j]; __syncthreads();
  for (int s = 64; s > 0; s >>= 1) {
    if (j < s) red[j] += red[j + s];
    __syncthreads();
  }
  if (j == 0) out[b] = 1.f / (1.f + __expf(-(red[0] + rb3[0])));
}

// ---------------------------------------------------------------------------
extern "C" void kernel_launch(void* const* d_in, const int* in_sizes, int n_in,
                              void* d_out, int out_size, void* d_ws, size_t ws_size,
                              hipStream_t stream)
{
  const float* demo   = (const float*)d_in[0];
  const float* times  = (const float*)d_in[1];
  const float* values = (const float*)d_in[2];
  const int*   meas   = (const int*)d_in[3];
  /* d_in[4] segment_ids: structure known statically */
  const float* ts     = (const float*)d_in[5];
  const float* dW1 = (const float*)d_in[6];  const float* db1 = (const float*)d_in[7];
  const float* dW2 = (const float*)d_in[8];  const float* db2 = (const float*)d_in[9];
  const float* pW0 = (const float*)d_in[10]; const float* pb0 = (const float*)d_in[11];
  const float* pW1 = (const float*)d_in[12]; const float* pb1 = (const float*)d_in[13];
  const float* pW2 = (const float*)d_in[14]; const float* pb2 = (const float*)d_in[15];
  const float* pW3 = (const float*)d_in[16]; const float* pb3 = (const float*)d_in[17];
  const float* qW0 = (const float*)d_in[18]; const float* qb0 = (const float*)d_in[19];
  const float* qW1 = (const float*)d_in[20]; const float* qb1 = (const float*)d_in[21];
  const float* qW2 = (const float*)d_in[22]; const float* qb2 = (const float*)d_in[23];
  const float* Wk  = (const float*)d_in[24];
  const float* Wq  = (const float*)d_in[25];
  const float* rW0 = (const float*)d_in[26]; const float* rb0 = (const float*)d_in[27];
  const float* rW1 = (const float*)d_in[28]; const float* rb1 = (const float*)d_in[29];
  const float* rW2 = (const float*)d_in[30]; const float* rb2 = (const float*)d_in[31];
  const float* rW3 = (const float*)d_in[32]; const float* rb3 = (const float*)d_in[33];

  char* ws = (char*)d_ws;
  _Float16* wAll   = (_Float16*)(ws + 0);         // 196608 B (phi 114688 + psi 81920)
  float*    u48p   = (float*)(ws + 196608);       // 1024 B
  float*    u128   = (float*)(ws + 197632);       // 2048 B
  float*    aggdot = (float*)(ws + 199680);       // 1024 B
  float*    offv   = (float*)(ws + 200704);       // 1024 B
  float*    invs   = (float*)(ws + 201728);       // 1024 B
  float*    segsum = (float*)(ws + 202752);       // 32768 B
  float*    aggout = (float*)(ws + 235520);       // 131072 B
  float*    cdot   = (float*)(ws + 366592);       // N*16  = 4195328 B
  _Float16* feat   = (_Float16*)(ws + 4561920);   // N*128 = 33562624 B
  _Float16* enc    = (_Float16*)(ws + 38124544);  // N*256 = 67125248 B

  hipMemsetAsync(segsum, 0, 64 * 128 * sizeof(float), stream);
  hipMemsetAsync(aggout, 0, 64 * 512 * sizeof(float), stream);

  prep_kernel<<<64, 256, 0, stream>>>(pW0, pW1, pW2, pW3, qW0, qW1, qW2,
                                      Wk, Wq, wAll, u48p, u128);
  feat_kernel<<<1024, 256, 0, stream>>>(times, values, meas, ts, u48p, feat, cdot);
  demo_kernel<<<64, 128, 0, stream>>>(demo, dW1, db1, dW2, db2,
                                      pW0, pb0, pW1, pb1, pW2, pb2, pW3, pb3,
                                      qW0, qb0, qW1, qb1, qW2, qb2,
                                      u48p, feat, enc, cdot, segsum);
  mlp_kernel<<<2048, 256, 229376, stream>>>(feat, wAll, pb0, pb1, pb2, pb3,
                                            qb0, qb1, qb2, enc, segsum);
  aggdot_kernel<<<1, 256, 0, stream>>>(segsum, u128, aggdot);
  softstat_kernel<<<64, 256, 0, stream>>>(cdot, aggdot, offv, invs);
  aggregate_kernel<<<512, 256, 0, stream>>>(enc, cdot, offv, invs, aggout);
  rho_kernel<<<64, 128, 0, stream>>>(aggout, rW0, rb0, rW1, rb1, rW2, rb2,
                                     rW3, rb3, (float*)d_out);
}
